// DiscreteContinuousConvS2_43550968382041
// MI455X (gfx1250) — matrix operand order repaired and checked by reading, not compiled, not run on hardware
//
#include <hip/hip_runtime.h>
#include <hip/hip_bf16.h>

// ---------------------------------------------------------------------------
// DISCO S2 conv for MI455X (gfx1250, wave32, WMMA).
//   Stage 1 (sparse gather-accumulate) -> y[(c*9+k)][p] tile in LDS (f32)
//   Stage 2 (channel mixing)           -> v_wmma_f32_16x16x32_bf16
// Weights are repacked once per launch into the ISA 16-bit A-matrix (16x32)
// per-lane fragment layout so stage-2 A loads are contiguous 32B per lane.
// ---------------------------------------------------------------------------

#define NLAT 361
#define NLON 720
#define KK 9
#define NNZ 65536
#define NB 2
#define NC 64          // CIN == COUT == 64
#define CK 576         // NC * KK
#define PT 16          // p-tile width (WMMA N)
#define KSTEPS 18      // CK / 32

typedef __attribute__((ext_vector_type(16))) __bf16 v16bf;
typedef __attribute__((ext_vector_type(8)))  float  v8f;

// ------------------------------- prepass -----------------------------------

__global__ void k_zero_counts(int* counts) {
    int i = blockIdx.x * blockDim.x + threadIdx.x;
    if (i < NLAT) counts[i] = 0;
}

__global__ void k_count(const int* __restrict__ row_idx, int* __restrict__ counts) {
    int i = blockIdx.x * blockDim.x + threadIdx.x;
    if (i < NNZ) atomicAdd(&counts[row_idx[i]], 1);
}

__global__ void k_scan(const int* __restrict__ counts, int* __restrict__ offs) {
    if (threadIdx.x == 0 && blockIdx.x == 0) {
        int a = 0;
        for (int i = 0; i < NLAT; ++i) { offs[i] = a; a += counts[i]; }
        offs[NLAT] = a;
    }
}

// One wave per output row: deterministic ballot-ordered compaction of nz
// records into CSR order.  Record: {lat_in, lon_in, k, bitcast(val)}.
__global__ void k_compact(const int* __restrict__ ker_idx,
                          const int* __restrict__ row_idx,
                          const int* __restrict__ col_idx,
                          const float* __restrict__ vals,
                          const int* __restrict__ offs,
                          int4* __restrict__ sorted) {
    int wid  = (blockIdx.x * blockDim.x + threadIdx.x) >> 5;
    int lane = threadIdx.x & 31;
    if (wid >= NLAT) return;
    int base = offs[wid];
    for (int i0 = 0; i0 < NNZ; i0 += 32) {          // NNZ % 32 == 0
        int i = i0 + lane;
        bool m = (row_idx[i] == wid);
        unsigned mask = __builtin_amdgcn_ballot_w32(m);
        if (m) {
            int pos = base + __popc(mask & ((1u << lane) - 1u));
            int c = col_idx[i];
            int4 rec;
            rec.x = c / NLON;                        // lat_in
            rec.y = c % NLON;                        // lon_in
            rec.z = ker_idx[i];                      // k
            rec.w = __float_as_int(vals[i]);         // psi value
            sorted[pos] = rec;
        }
        base += __popc(mask);
    }
}

// Repack weight (COUT,CIN,K) f32 -> bf16 A-fragments.
// Fragment (t = cout tile 0..3, s = kstep 0..17): 32 lanes x 16 bf16,
// lane-contiguous.  ISA 16-bit A 16x32 layout:
//   M = lane%16,  K = (e<8 ? e : e+8) + 8*(lane/16),  e = element 0..15
__global__ void k_pack_w(const float* __restrict__ weight, __bf16* __restrict__ Wb) {
    int idx = blockIdx.x * blockDim.x + threadIdx.x;   // < 4*18*512 = 36864
    if (idx >= 4 * KSTEPS * 512) return;
    int ts   = idx >> 9;            // frag id 0..71
    int t    = ts / KSTEPS;
    int s    = ts % KSTEPS;
    int lane = (idx >> 4) & 31;
    int e    = idx & 15;
    int M    = lane & 15;
    int Kl   = (e < 8 ? e : e + 8) + ((lane >> 4) << 3);
    int ck   = s * 32 + Kl;                         // < 576
    int cout = t * 16 + M;
    int cin  = ck / KK;
    int kk   = ck % KK;
    Wb[idx] = (__bf16)weight[(cout * NC + cin) * KK + kk];
}

// ------------------------------ main kernel --------------------------------
// grid = (NLON/PT, NLAT, NB), block = 256 threads (8 waves).
__global__ __launch_bounds__(256)
void k_main(const float* __restrict__ x,
            const float* __restrict__ bias,
            const int* __restrict__ offs,
            const int4* __restrict__ sorted,
            const __bf16* __restrict__ Wb,
            float* __restrict__ out) {
    __shared__ float s_acc[CK * PT];      // y[(c*9+k)][p], f32, 36 KB
    __shared__ float s_red[4 * 32 * 8];   // cross-wave GEMM partials, 4 KB

    const int p0  = blockIdx.x * PT;
    const int lo  = blockIdx.y;
    const int b   = blockIdx.z;
    const int tid = threadIdx.x;

    for (int i = tid; i < CK * PT; i += 256) s_acc[i] = 0.0f;
    __syncthreads();

    // ---- stage 1: sparse gather-accumulate into the y tile ----------------
    // thread owns p = tid%16 and channels c = tid/16 + 16j (exclusive LDS
    // slots -> plain ds read-modify-write, no atomics, conflict-free).
    {
        const int p  = tid & 15;
        const int cb = tid >> 4;
        const int start = offs[lo];
        const int end   = offs[lo + 1];
        const float* xb = x + (size_t)b * NC * NLAT * NLON;
        for (int r = start; r < end; ++r) {
            int4 rec = sorted[r];
            __builtin_prefetch(&sorted[r + 1], 0, 1);   // global_prefetch_b8
            const float val = __int_as_float(rec.w);
            int lon = rec.y + p0 + p;
            if (lon >= NLON) lon -= NLON;               // max 1438 -> one sub
            const float* xr = xb + (size_t)rec.x * NLON + lon;
#pragma unroll
            for (int j = 0; j < 4; ++j) {
                const int c = cb + 16 * j;
                const float xv = xr[(size_t)c * NLAT * NLON];
                const int ai = ((c * KK + rec.z) << 4) + p;
                s_acc[ai] += val * xv;                  // exclusive owner
            }
        }
    }
    __syncthreads();

    // ---- stage 2: out[cout,p] = W[cout,ck] @ y[ck,p] via bf16 WMMA --------
    // 8 waves: wave w -> cout tile t=w%4, K half = w/4 (9 ksteps each).
    {
        const int w    = tid >> 5;
        const int lane = tid & 31;
        const int t    = w & 3;
        const int half = w >> 2;
        const int nl   = lane & 15;                 // N (p within tile)
        const int kb   = (lane >> 4) << 4;          // B-matrix K base per lane

        v8f acc = {};
        const int s_beg = half * 9;
#pragma unroll
        for (int si = 0; si < 9; ++si) {
            const int s = s_beg + si;
            // A fragment: contiguous 32 B per lane from the packed weights
            v16bf a = *(const v16bf*)(Wb + (((t * KSTEPS + s) * 32 + lane) << 4));
            // B fragment: gather 16 f32 from the LDS y tile, convert to bf16
            // B (32x16) layout: N = lane%16, K = (e<8 ? e : e+8) + 8*(lane/16)
            v16bf bm;
#pragma unroll
            for (int e = 0; e < 16; ++e)
                bm[e] = (__bf16)s_acc[((s * 32 + (kb >> 1) + (e < 8 ? e : e + 8)) << 4) + nl];
            acc = __builtin_amdgcn_wmma_f32_16x16x32_bf16(
                false, a, false, bm, (short)0, acc, false, false);
        }

        if (half) {
#pragma unroll
            for (int rr = 0; rr < 8; ++rr)
                s_red[(t * 32 + lane) * 8 + rr] = acc[rr];
        }
        __syncthreads();
        if (!half) {
            const int mb = (lane >> 4) << 3;        // C/D layout: M = rr + 8*(lane/16)
#pragma unroll
            for (int rr = 0; rr < 8; ++rr) {
                const int cout = t * 16 + mb + rr;
                float v = acc[rr] + s_red[(t * 32 + lane) * 8 + rr] + bias[cout];
                out[(((size_t)b * NC + cout) * NLAT + lo) * NLON + p0 + nl] = v;
            }
        }
    }
}

// ------------------------------- launcher ----------------------------------

extern "C" void kernel_launch(void* const* d_in, const int* in_sizes, int n_in,
                              void* d_out, int out_size, void* d_ws, size_t ws_size,
                              hipStream_t stream) {
    const float* x        = (const float*)d_in[0];
    const float* psi_vals = (const float*)d_in[1];
    const float* weight   = (const float*)d_in[2];
    const float* bias     = (const float*)d_in[3];
    const int*   ker_idx  = (const int*)d_in[4];
    const int*   row_idx  = (const int*)d_in[5];
    const int*   col_idx  = (const int*)d_in[6];
    float* out = (float*)d_out;

    // workspace layout
    char* ws = (char*)d_ws;
    int*    counts = (int*)ws;                                   // 361 ints
    int*    offs   = (int*)(ws + 2048);                          // 362 ints
    int4*   sorted = (int4*)(ws + 4096);                         // 1 MB
    __bf16* Wb     = (__bf16*)(ws + 4096 + (size_t)NNZ * 16);    // 72 KB

    k_zero_counts<<<2, 256, 0, stream>>>(counts);
    k_count<<<NNZ / 256, 256, 0, stream>>>(row_idx, counts);
    k_scan<<<1, 32, 0, stream>>>(counts, offs);
    k_compact<<<(NLAT * 32 + 255) / 256, 256, 0, stream>>>(
        ker_idx, row_idx, col_idx, psi_vals, offs, sorted);
    k_pack_w<<<(4 * KSTEPS * 512 + 255) / 256, 256, 0, stream>>>(weight, Wb);

    dim3 grid(NLON / PT, NLAT, NB);   // 45 x 361 x 2 blocks
    k_main<<<grid, 256, 0, stream>>>(x, bias, offs, sorted, Wb, out);
}
